// Self_Attn_70798240907901
// MI455X (gfx1250) — matrix operand order repaired and checked by reading, not compiled, not run on hardware
//
#include <hip/hip_runtime.h>
#include <hip/hip_bf16.h>

// ---------------------------------------------------------------------------
// Self-attention (SAGAN) fused pipeline for MI455X / gfx1250.
//   B=4, C=256, N=4096, Cqk=32.  bf16 WMMA everywhere, flash-style online
//   softmax (no NxN matrix in HBM), TDM async tensor loads double-buffered
//   into LDS, wave32 fragment layouts per CDNA5 ISA 7.12.2.
// ---------------------------------------------------------------------------

#define B_   4
#define C_   256
#define N_   4096
#define DQK  32
#define LOG2E 1.4426950408889634f

typedef __attribute__((ext_vector_type(16))) __bf16      v16bf;
typedef __attribute__((ext_vector_type(8)))  float       v8f;
typedef __attribute__((ext_vector_type(4)))  float       v4f;
typedef __attribute__((ext_vector_type(4)))  unsigned    v4u;
typedef __attribute__((ext_vector_type(8)))  int         v8i;
typedef __attribute__((ext_vector_type(4)))  int         v4i;

struct Frag32B { v4u lo, hi; };

static __device__ __forceinline__ v16bf make_frag(v4u lo, v4u hi) {
  Frag32B f{lo, hi};
  return __builtin_bit_cast(v16bf, f);
}
// Load a 16-element bf16 fragment from two 16-byte chunks.
static __device__ __forceinline__ v16bf load_frag(const unsigned short* p0,
                                                  const unsigned short* p1) {
  return make_frag(*(const v4u*)p0, *(const v4u*)p1);
}
static __device__ __forceinline__ unsigned short f2bfu(float f) {
  __bf16 h = (__bf16)f;
  return __builtin_bit_cast(unsigned short, h);
}
static __device__ __forceinline__ v8f wmma_bf16(v16bf a, v16bf b, v8f c) {
  // D(f32 16x16) = A(16x32 bf16) x B(32x16 bf16) + C
  return __builtin_amdgcn_wmma_f32_16x16x32_bf16(false, a, false, b,
                                                 (short)0, c, false, false);
}

// Butterfly reductions across the 16 lanes of a wave-half (C-matrix rows live
// entirely inside one half).  ds_swizzle group-of-32: offset = xor<<10 | 0x1f.
#define SWZ_XOR_F(x, m) \
  __int_as_float(__builtin_amdgcn_ds_swizzle(__float_as_int(x), (((m) << 10) | 0x1f)))

static __device__ __forceinline__ float redmax_half(float x) {
  x = fmaxf(x, SWZ_XOR_F(x, 1));
  x = fmaxf(x, SWZ_XOR_F(x, 2));
  x = fmaxf(x, SWZ_XOR_F(x, 4));
  x = fmaxf(x, SWZ_XOR_F(x, 8));
  return x;
}
static __device__ __forceinline__ float redsum_half(float x) {
  x += SWZ_XOR_F(x, 1);
  x += SWZ_XOR_F(x, 2);
  x += SWZ_XOR_F(x, 4);
  x += SWZ_XOR_F(x, 8);
  return x;
}

#if __has_builtin(__builtin_amdgcn_tensor_load_to_lds)
#define USE_TDM 1
#else
#define USE_TDM 0
#endif

#if USE_TDM
union G1u { unsigned long long q[4]; v8i v; };
// 2-D tensor-tile DMA: tile (w x h) of 2-byte elements, row stride in elems.
// D# fields packed per CDNA5 ISA §8.3 (group0) / §8.4 (group1).
// Toolchain here exposes the 6-arg builtin:
//   (uint32x4 g0, int32x8 g1, int32x4 g2, int32x4 g3, int32x8 g23hi, i32 cpol)
// groups 2/3 are only needed for >=3-D tensors -> zeros for 2-D tiles.
static __device__ __forceinline__ void tdm_load2d(unsigned lds_off, const void* g,
                                                  unsigned w, unsigned h,
                                                  unsigned long long stride) {
  unsigned long long ga = (unsigned long long)g;
  v4u g0;
  g0[0] = 1u;                                                   // count=1
  g0[1] = lds_off;                                              // lds_addr
  g0[2] = (unsigned)ga;                                         // global_addr lo
  g0[3] = (unsigned)((ga >> 32) & 0x01ffffffu) | 0x80000000u;   // addr hi | type=2
  G1u g1;
  unsigned long long w64 = w, h64 = h;
  g1.q[0] = (1ull << 16) | (w64 << 48);                         // data_size=2B; dim0 lo
  g1.q[1] = (w64 >> 16) | (h64 << 16) | (w64 << 48);            // dim0 hi; dim1; tile_dim0
  g1.q[2] = h64 | ((stride & 0xffffffffull) << 32);             // tile_dim1; stride0 lo
  g1.q[3] = (stride >> 32) & 0xffffull;                         // stride0 hi; stride1=0
  v4i z4 = {0, 0, 0, 0};
  v8i z8 = {0, 0, 0, 0, 0, 0, 0, 0};
  __builtin_amdgcn_tensor_load_to_lds(g0, g1.v, z4, z4, z8, 0);
}
#endif

// ---------------------------------------------------------------------------
// K0: weights f32 -> bf16
__global__ void prepw_kernel(const float* __restrict__ Wq, const float* __restrict__ Wk,
                             const float* __restrict__ Wv, const float* __restrict__ Wo,
                             unsigned short* __restrict__ wq, unsigned short* __restrict__ wk,
                             unsigned short* __restrict__ wv, unsigned short* __restrict__ wo) {
  const int i = blockIdx.x * 256 + threadIdx.x;          // grid covers 65536
  if (i < DQK * C_) { wq[i] = f2bfu(Wq[i]); wk[i] = f2bfu(Wk[i]); }
  wv[i] = f2bfu(Wv[i]);
  wo[i] = f2bfu(Wo[i]);
}

// K1: x [B][C][N] f32 -> xT [B][N][C] bf16 (LDS-tiled transpose)
__global__ void xt_kernel(const float* __restrict__ x, unsigned short* __restrict__ xT) {
  __shared__ float tile[32][33];
  const int b = blockIdx.z;
  const int n0 = blockIdx.x * 32, c0 = blockIdx.y * 32;
  const int tx = threadIdx.x, ty = threadIdx.y;          // 32 x 8
  #pragma unroll
  for (int rr = 0; rr < 4; ++rr) {
    const int r = ty + rr * 8;
    tile[r][tx] = x[((size_t)b * C_ + c0 + r) * N_ + n0 + tx];
  }
  __syncthreads();
  #pragma unroll
  for (int rr = 0; rr < 4; ++rr) {
    const int r = ty + rr * 8;
    xT[((size_t)b * N_ + n0 + r) * C_ + c0 + tx] = f2bfu(tile[tx][r]);
  }
}

// K2: q_t/k_t [B][N][32] = xT x W^T + bias.  8 waves = 4 n-tiles x {q,k}.
__global__ __launch_bounds__(256) void qk_kernel(
    const unsigned short* __restrict__ xT,
    const unsigned short* __restrict__ wq, const unsigned short* __restrict__ wk,
    const float* __restrict__ bq, const float* __restrict__ bk,
    unsigned short* __restrict__ q_t, unsigned short* __restrict__ k_t) {
  const int tid = threadIdx.x, wid = tid >> 5, lane = tid & 31;
  const int h = lane >> 4, l16 = lane & 15;
  const int b = blockIdx.y;
  const int nt = blockIdx.x * 4 + (wid >> 1);
  const int isK = wid & 1;
  const unsigned short* W = isK ? wk : wq;
  const float* bias = isK ? bk : bq;
  unsigned short* out = isK ? k_t : q_t;

  const unsigned short* xrow = xT + ((size_t)b * N_ + nt * 16 + l16) * C_;
  v8f zero = {};
  v8f acc0 = zero, acc1 = zero;
  #pragma unroll
  for (int ks = 0; ks < 8; ++ks) {
    v16bf a = load_frag(xrow + ks * 32 + 8 * h, xrow + ks * 32 + 16 + 8 * h);
    const unsigned short* wr0 = W + (size_t)l16 * C_ + ks * 32 + 8 * h;
    const unsigned short* wr1 = W + (size_t)(16 + l16) * C_ + ks * 32 + 8 * h;
    acc0 = wmma_bf16(a, load_frag(wr0, wr0 + 16), acc0);
    acc1 = wmma_bf16(a, load_frag(wr1, wr1 + 16), acc1);
  }
  const float b0 = bias[l16], b1 = bias[16 + l16];
  #pragma unroll
  for (int v = 0; v < 8; ++v) {
    const int n = nt * 16 + v + 8 * h;
    out[((size_t)b * N_ + n) * DQK + l16]      = f2bfu(acc0[v] + b0);
    out[((size_t)b * N_ + n) * DQK + 16 + l16] = f2bfu(acc1[v] + b1);
  }
}

// K3: v [B][C][N] = Wv x + bv.  One 16c x 16n tile per wave.
__global__ __launch_bounds__(256) void v_kernel(
    const unsigned short* __restrict__ xT, const unsigned short* __restrict__ wv,
    const float* __restrict__ bv, unsigned short* __restrict__ vout) {
  const int tid = threadIdx.x, wid = tid >> 5, lane = tid & 31;
  const int h = lane >> 4, l16 = lane & 15;
  const int b = blockIdx.z, ct = blockIdx.y;
  const int nt = blockIdx.x * 8 + wid;
  const unsigned short* wrow = wv + (size_t)(ct * 16 + l16) * C_;
  const unsigned short* xrow = xT + ((size_t)b * N_ + nt * 16 + l16) * C_;
  v8f acc = {};
  #pragma unroll
  for (int ks = 0; ks < 8; ++ks) {
    v16bf a  = load_frag(wrow + ks * 32 + 8 * h, wrow + ks * 32 + 16 + 8 * h);
    v16bf bb = load_frag(xrow + ks * 32 + 8 * h, xrow + ks * 32 + 16 + 8 * h);
    acc = wmma_bf16(a, bb, acc);
  }
  #pragma unroll
  for (int v = 0; v < 8; ++v) {
    const int c = ct * 16 + v + 8 * h;
    vout[((size_t)b * C_ + c) * N_ + nt * 16 + l16] = f2bfu(acc[v] + bv[c]);
  }
}

// K4: fused flash attention.  Block = 8 waves = 4 i-tiles x 2 channel halves.
//     j-loop in chunks of 32 with double-buffered TDM loads of q & V into LDS.
__global__ __launch_bounds__(256) void attn_kernel(
    const unsigned short* __restrict__ q_t, const unsigned short* __restrict__ k_t,
    const unsigned short* __restrict__ vmat, unsigned short* __restrict__ OT) {
  __shared__ __align__(16) unsigned short qbuf[2][32 * DQK];   // [j][d]
  __shared__ __align__(16) unsigned short vbuf[2][C_ * 32];    // [c][j]
  __shared__ __align__(16) float          pbuf[8][16 * 32];    // per-wave P staging

  const int tid = threadIdx.x;
  const int wid = tid >> 5, lane = tid & 31;
  const int h = lane >> 4, l16 = lane & 15;
  const int b = blockIdx.y;
  const int itile = wid >> 1, chalf = wid & 1;
  const int i0 = blockIdx.x * 64 + itile * 16;

  const unsigned short* qb_g = q_t + (size_t)b * N_ * DQK;
  const unsigned short* vb_g = vmat + (size_t)b * C_ * N_;

  // persistent A-fragment of K for this wave's 16 rows
  const unsigned short* krow = k_t + ((size_t)b * N_ + (i0 + l16)) * DQK;
  v16bf kf = load_frag(krow + 8 * h, krow + 16 + 8 * h);

  v8f zero = {};
  v8f acc[8];
  #pragma unroll
  for (int i = 0; i < 8; ++i) acc[i] = zero;
  float mrow[8], lrow[8];
  #pragma unroll
  for (int i = 0; i < 8; ++i) { mrow[i] = -3.0e38f; lrow[i] = 0.0f; }

  const int T = N_ / 32;

  auto load_chunk = [&](int tc) {
    const int j0 = tc * 32;
    const int bb = tc & 1;
#if USE_TDM
    if (wid == 0) {
      tdm_load2d((unsigned)(unsigned long long)(void*)&qbuf[bb][0],
                 qb_g + (size_t)j0 * DQK, DQK, 32, DQK);
      tdm_load2d((unsigned)(unsigned long long)(void*)&vbuf[bb][0],
                 vb_g + j0, 32, C_, (unsigned long long)N_);
    }
#else
    if (tid < 128)
      ((v4u*)&qbuf[bb][0])[tid] = *(const v4u*)(qb_g + (size_t)j0 * DQK + tid * 8);
    #pragma unroll
    for (int k = 0; k < 4; ++k) {
      const int idx = tid + k * 256;                  // 0..1023
      const int c = idx >> 2, part = idx & 3;
      ((v4u*)&vbuf[bb][0])[idx] = *(const v4u*)(vb_g + (size_t)c * N_ + j0 + part * 8);
    }
#endif
  };

  load_chunk(0);
  for (int t = 0; t < T; ++t) {
    const int bi = t & 1;
    if (t + 1 < T) load_chunk(t + 1);
#if USE_TDM
    if (wid == 0) {
      if (t + 1 < T) __builtin_amdgcn_s_wait_tensorcnt(2);   // chunk t complete
      else           __builtin_amdgcn_s_wait_tensorcnt(0);
    }
#endif
    __syncthreads();

    // ---- S tile (16 x 32) = K^T Q ----
    const unsigned short* qb = &qbuf[bi][0];
    v16bf qf0 = load_frag(qb + l16 * DQK + 8 * h, qb + l16 * DQK + 16 + 8 * h);
    v16bf qf1 = load_frag(qb + (16 + l16) * DQK + 8 * h,
                          qb + (16 + l16) * DQK + 16 + 8 * h);
    v8f s0 = wmma_bf16(kf, qf0, zero);
    v8f s1 = wmma_bf16(kf, qf1, zero);

    // ---- online softmax over this 32-wide j chunk ----
    float al[8];
    float* pw = &pbuf[wid][0];
    #pragma unroll
    for (int v = 0; v < 8; ++v) {
      const float cm = redmax_half(fmaxf(s0[v], s1[v]));
      const float mn = fmaxf(mrow[v], cm);
      const float a  = exp2f((mrow[v] - mn) * LOG2E);
      const float p0 = exp2f((s0[v] - mn) * LOG2E);
      const float p1 = exp2f((s1[v] - mn) * LOG2E);
      lrow[v] = lrow[v] * a + redsum_half(p0 + p1);
      mrow[v] = mn;
      al[v] = a;
      const int m = v + 8 * h;
      pw[m * 32 + l16]      = p0;        // conflict-free: 16 consecutive dwords
      pw[m * 32 + 16 + l16] = p1;
    }
    #pragma unroll
    for (int ct = 0; ct < 8; ++ct) {
      #pragma unroll
      for (int v = 0; v < 8; ++v) acc[ct][v] *= al[v];
    }

    // ---- re-read P as bf16 A-fragment (same-wave DS ops are in order) ----
    const float* pr = &pbuf[wid][0];
    v4f t0 = *(const v4f*)(pr + l16 * 32 + 8 * h);
    v4f t1 = *(const v4f*)(pr + l16 * 32 + 8 * h + 4);
    v4f t2 = *(const v4f*)(pr + l16 * 32 + 16 + 8 * h);
    v4f t3 = *(const v4f*)(pr + l16 * 32 + 16 + 8 * h + 4);
    v16bf pf;
    #pragma unroll
    for (int e = 0; e < 4; ++e) {
      pf[e]      = (__bf16)t0[e];
      pf[e + 4]  = (__bf16)t1[e];
      pf[e + 8]  = (__bf16)t2[e];
      pf[e + 12] = (__bf16)t3[e];
    }

    // ---- O += P x V^T over this wave's 128 channels ----
    #pragma unroll
    for (int ct = 0; ct < 8; ++ct) {
      const int c = chalf * 128 + ct * 16 + l16;
      const unsigned short* vp = &vbuf[bi][c * 32 + 8 * h];
      acc[ct] = wmma_bf16(pf, load_frag(vp, vp + 16), acc[ct]);
    }
    __syncthreads();          // release buf[bi] for the t+2 prefetch
  }

  // ---- epilogue: normalize and store O^T [B][N][C] bf16 ----
  #pragma unroll
  for (int v = 0; v < 8; ++v) lrow[v] = 1.0f / lrow[v];
  #pragma unroll
  for (int ct = 0; ct < 8; ++ct) {
    const int c = chalf * 128 + ct * 16 + l16;
    #pragma unroll
    for (int v = 0; v < 8; ++v) {
      const int i = i0 + v + 8 * h;
      OT[((size_t)b * N_ + i) * C_ + c] = f2bfu(acc[ct][v] * lrow[v]);
    }
  }
}

// K5: out = x + gamma * (Wo . O + bo)
__global__ __launch_bounds__(256) void out_kernel(
    const unsigned short* __restrict__ OT, const unsigned short* __restrict__ wo,
    const float* __restrict__ bo, const float* __restrict__ x,
    const float* __restrict__ gamma, float* __restrict__ out) {
  const int tid = threadIdx.x, wid = tid >> 5, lane = tid & 31;
  const int h = lane >> 4, l16 = lane & 15;
  const int b = blockIdx.z, cot = blockIdx.y;
  const int nt = blockIdx.x * 8 + wid;
  const unsigned short* wrow = wo + (size_t)(cot * 16 + l16) * C_;
  const unsigned short* orow = OT + ((size_t)b * N_ + nt * 16 + l16) * C_;
  v8f acc = {};
  #pragma unroll
  for (int ks = 0; ks < 8; ++ks) {
    v16bf a  = load_frag(wrow + ks * 32 + 8 * h, wrow + ks * 32 + 16 + 8 * h);
    v16bf bb = load_frag(orow + ks * 32 + 8 * h, orow + ks * 32 + 16 + 8 * h);
    acc = wmma_bf16(a, bb, acc);
  }
  const float g = gamma[0];
  #pragma unroll
  for (int v = 0; v < 8; ++v) {
    const int co = cot * 16 + v + 8 * h;
    const size_t idx = ((size_t)b * C_ + co) * N_ + nt * 16 + l16;
    out[idx] = x[idx] + g * (acc[v] + bo[co]);    // coalesced across lanes
  }
}

// ---------------------------------------------------------------------------
extern "C" void kernel_launch(void* const* d_in, const int* in_sizes, int n_in,
                              void* d_out, int out_size, void* d_ws, size_t ws_size,
                              hipStream_t stream) {
  (void)in_sizes; (void)n_in; (void)out_size; (void)ws_size;
  const float* x     = (const float*)d_in[0];
  const float* Wq    = (const float*)d_in[1];
  const float* bq    = (const float*)d_in[2];
  const float* Wk    = (const float*)d_in[3];
  const float* bk    = (const float*)d_in[4];
  const float* Wv    = (const float*)d_in[5];
  const float* bv    = (const float*)d_in[6];
  const float* Wo    = (const float*)d_in[7];
  const float* bo    = (const float*)d_in[8];
  const float* gamma = (const float*)d_in[9];
  float* out = (float*)d_out;

  char* ws = (char*)d_ws;
  unsigned short* xT  = (unsigned short*)(ws);                    // 8 MiB (dead after K3)
  unsigned short* q_t = (unsigned short*)(ws + (8ull  << 20));    // 1 MiB
  unsigned short* k_t = (unsigned short*)(ws + (9ull  << 20));    // 1 MiB
  unsigned short* vmt = (unsigned short*)(ws + (10ull << 20));    // 8 MiB
  unsigned short* wq  = (unsigned short*)(ws + (18ull << 20));
  unsigned short* wk  = wq + DQK * C_;
  unsigned short* wv  = wk + DQK * C_;
  unsigned short* wo  = wv + C_ * C_;
  unsigned short* OT  = xT;                                       // reuse xT region

  prepw_kernel<<<dim3(256), dim3(256), 0, stream>>>(Wq, Wk, Wv, Wo, wq, wk, wv, wo);
  xt_kernel<<<dim3(N_ / 32, C_ / 32, B_), dim3(32, 8), 0, stream>>>(x, xT);
  qk_kernel<<<dim3(N_ / 64, B_), dim3(256), 0, stream>>>(xT, wq, wk, bq, bk, q_t, k_t);
  v_kernel<<<dim3(N_ / 128, C_ / 16, B_), dim3(256), 0, stream>>>(xT, wv, bv, vmt);
  attn_kernel<<<dim3(N_ / 64, B_), dim3(256), 0, stream>>>(q_t, k_t, vmt, OT);
  out_kernel<<<dim3(N_ / 128, C_ / 16, B_), dim3(256), 0, stream>>>(OT, wo, bo, x, gamma, out);
}
